// CustomGRU_70153995813516
// MI455X (gfx1250) — matrix operand order repaired and checked by reading, not compiled, not run on hardware
//
#include <hip/hip_runtime.h>
#include <hip/hip_bf16.h>
#include <cstdint>

// ---------------------------------------------------------------------------
// CDNA5 (gfx1250) stacked-GRU.
//  - bf16 operands + v_wmma_f32_16x16x32_bf16 (f32 accumulate)
//  - per-layer: one big parallel input-projection GEMM (M = B*T) with
//    64x32 wave tiles (8 WMMA per 6 fragment loads),
//    then T sequential (zr-GEMM, n-GEMM) step kernels with fused gate math,
//    state update and masked time-sum (32x32 wave tiles for parallelism).
// ---------------------------------------------------------------------------

typedef __attribute__((ext_vector_type(16))) __bf16 bf16x16;
typedef __attribute__((ext_vector_type(8)))  float  f32x8;

#define WMMA_BF16(A_, B_, C_) \
  __builtin_amdgcn_wmma_f32_16x16x32_bf16(false, (A_), false, (B_), (short)0, (C_), false, false)

// A fragment: 16(M) x 32(K), 16-bit.  ISA 7.12.2: lane L holds row (L&15);
// elements 0..7 <- K = 8g..8g+7, elements 8..15 <- K = 16+8g..16+8g+7 (g = L>>4).
__device__ __forceinline__ bf16x16 load_a_frag(const __bf16* A, int lda, int row0,
                                               int kb, int lane) {
  const int r = row0 + (lane & 15);
  const int g = lane >> 4;
  const __bf16* p = A + (size_t)r * lda + kb + 8 * g;
  union { bf16x16 v; uint4 u[2]; } f;
  f.u[0] = *(const uint4*)(p);        // K = kb+8g .. +7
  f.u[1] = *(const uint4*)(p + 16);   // K = kb+16+8g .. +7
  return f.v;
}

// B fragment: 32(K) x 16(N), 16-bit, B = W^T with W [N,K] row-major.
// Lane L holds column (L&15); elements 0..7 <- K = 8g..8g+7, 8..15 <- K = 16+8g..16+8g+7 (g = L>>4).
__device__ __forceinline__ bf16x16 load_b_frag(const __bf16* W, int ldw, int col0,
                                               int kb, int lane) {
  const int c = col0 + (lane & 15);
  const int g = lane >> 4;
  const __bf16* p = W + (size_t)c * ldw + kb + 8 * g;
  union { bf16x16 v; uint4 u[2]; } f;
  f.u[0] = *(const uint4*)(p);        // K = kb+8g .. +7
  f.u[1] = *(const uint4*)(p + 16);   // K = kb+16+8g .. +7
  return f.v;
}

// One wave computes a 32x32 tile of C = A[M,K] * W[N,K]^T as 2x2 WMMA tiles.
__device__ __forceinline__ void gemm_wave_32x32(const __bf16* A, int lda,
                                                const __bf16* W, int ldw,
                                                int row0, int col0, int K, int lane,
                                                f32x8 acc[2][2]) {
  const f32x8 z8 = {0.f, 0.f, 0.f, 0.f, 0.f, 0.f, 0.f, 0.f};
  acc[0][0] = z8; acc[0][1] = z8; acc[1][0] = z8; acc[1][1] = z8;
  for (int kb = 0; kb < K; kb += 32) {
    bf16x16 a0 = load_a_frag(A, lda, row0,      kb, lane);
    bf16x16 a1 = load_a_frag(A, lda, row0 + 16, kb, lane);
    bf16x16 b0 = load_b_frag(W, ldw, col0,      kb, lane);
    bf16x16 b1 = load_b_frag(W, ldw, col0 + 16, kb, lane);
    acc[0][0] = WMMA_BF16(a0, b0, acc[0][0]);
    acc[0][1] = WMMA_BF16(a0, b1, acc[0][1]);
    acc[1][0] = WMMA_BF16(a1, b0, acc[1][0]);
    acc[1][1] = WMMA_BF16(a1, b1, acc[1][1]);
  }
}

// One wave computes a 64x32 tile as 4x2 WMMA tiles (8 WMMA per 6 frag loads).
__device__ __forceinline__ void gemm_wave_64x32(const __bf16* A, int lda,
                                                const __bf16* W, int ldw,
                                                int row0, int col0, int K, int lane,
                                                f32x8 acc[4][2]) {
  const f32x8 z8 = {0.f, 0.f, 0.f, 0.f, 0.f, 0.f, 0.f, 0.f};
  for (int i = 0; i < 4; ++i) { acc[i][0] = z8; acc[i][1] = z8; }
  for (int kb = 0; kb < K; kb += 32) {
    bf16x16 b0 = load_b_frag(W, ldw, col0,      kb, lane);
    bf16x16 b1 = load_b_frag(W, ldw, col0 + 16, kb, lane);
#pragma unroll
    for (int i = 0; i < 4; ++i) {
      bf16x16 a = load_a_frag(A, lda, row0 + 16 * i, kb, lane);
      acc[i][0] = WMMA_BF16(a, b0, acc[i][0]);
      acc[i][1] = WMMA_BF16(a, b1, acc[i][1]);
    }
  }
}

// f32 -> bf16 cast, 4 elements / thread (all array sizes are multiples of 4).
__global__ void __launch_bounds__(256)
cast_f32_to_bf16_x4(const float* __restrict__ src, __bf16* __restrict__ dst, int n4) {
  int i = blockIdx.x * blockDim.x + threadIdx.x;
  if (i < n4) {
    const float4 s = ((const float4*)src)[i];
    union { __bf16 h[4]; unsigned long long u; } o;
    o.h[0] = (__bf16)s.x; o.h[1] = (__bf16)s.y;
    o.h[2] = (__bf16)s.z; o.h[3] = (__bf16)s.w;
    ((unsigned long long*)dst)[i] = o.u;
  }
}

// Input projection: GI[M, N] = X[M, K] @ Wih[N, K]^T + bias[N]   (M = B*T)
// Block = 256 threads = 8 waves (2 along M x 4 along N) -> 128x128 block tile,
// each wave owns a 64x32 sub-tile (4x2 WMMA register tiling).
__global__ void __launch_bounds__(256)
gi_gemm_kernel(const __bf16* __restrict__ X, const __bf16* __restrict__ Wih,
               const float* __restrict__ bias, float* __restrict__ GI,
               int N, int K) {
  const int lane = threadIdx.x & 31;
  const int wave = threadIdx.x >> 5;
  const int row0 = blockIdx.y * 128 + (wave & 1) * 64;
  const int col0 = blockIdx.x * 128 + (wave >> 1) * 32;
  f32x8 acc[4][2];
  gemm_wave_64x32(X, K, Wih, K, row0, col0, K, lane, acc);
  const int nl = lane & 15, g = lane >> 4;
  for (int i = 0; i < 4; ++i)
    for (int j = 0; j < 2; ++j) {
      union { f32x8 v; float f[8]; } c; c.v = acc[i][j];
      const int n  = col0 + j * 16 + nl;
      const float bv = bias[n];
      for (int v = 0; v < 8; ++v) {
        const int m = row0 + i * 16 + g * 8 + v;
        GI[(size_t)m * N + n] = c.f[v] + bv;
      }
    }
}

// Step part 1: s = GI_t[:, 0:2H] + h @ [Uz;Ur]^T ; z=sigmoid, r=sigmoid,
// store z (f32) and r*h (bf16) for the n-gate GEMM.  M = B = 128 (one block row).
__global__ void __launch_bounds__(256)
zr_step_kernel(const __bf16* __restrict__ h_bf, const float* __restrict__ h_f,
               const __bf16* __restrict__ Whh, const float* __restrict__ GI_t,
               int ldGI, float* __restrict__ z_buf, __bf16* __restrict__ rh_bf,
               int H) {
  const int lane = threadIdx.x & 31;
  const int wave = threadIdx.x >> 5;
  const int row0 = (wave & 3) * 32;
  const int col0 = blockIdx.x * 64 + (wave >> 2) * 32;   // n in [0, 2H)
  f32x8 acc[2][2];
  gemm_wave_32x32(h_bf, H, Whh, H, row0, col0, H, lane, acc);
  const int nl = lane & 15, g = lane >> 4;
  for (int i = 0; i < 2; ++i)
    for (int j = 0; j < 2; ++j) {
      union { f32x8 v; float f[8]; } c; c.v = acc[i][j];
      const int n = col0 + j * 16 + nl;
      for (int v = 0; v < 8; ++v) {
        const int m = row0 + i * 16 + g * 8 + v;
        const float val = c.f[v] + GI_t[(size_t)m * ldGI + n];
        const float s = 1.0f / (1.0f + __expf(-val));
        if (n < H) {
          z_buf[m * H + n] = s;                                   // z gate
        } else {
          const int jj = n - H;                                   // r gate
          rh_bf[m * H + jj] = (__bf16)(s * h_f[m * H + jj]);      // r * h
        }
      }
    }
}

// Step part 2: n = tanh(GI_t[:,2H:3H] + (r*h) @ Un^T);
// h' = (1-z)*n + z*h ; out[b, off+j] += mask[b,t] * h' ; emit y (bf16) + h (f32/bf16).
__global__ void __launch_bounds__(256)
n_step_kernel(const __bf16* __restrict__ rh_bf, const __bf16* __restrict__ Un,
              const float* __restrict__ GIn_t, int ldGI,
              const float* __restrict__ z_buf, float* __restrict__ h_f,
              __bf16* __restrict__ h_bf, __bf16* __restrict__ y_bf,
              float* __restrict__ out, const float* __restrict__ mask,
              int t, int T, int H, int out_off, int outW) {
  const int lane = threadIdx.x & 31;
  const int wave = threadIdx.x >> 5;
  const int row0 = (wave & 3) * 32;
  const int col0 = blockIdx.x * 64 + (wave >> 2) * 32;   // n in [0, H)
  f32x8 acc[2][2];
  gemm_wave_32x32(rh_bf, H, Un, H, row0, col0, H, lane, acc);
  const int nl = lane & 15, g = lane >> 4;
  for (int i = 0; i < 2; ++i)
    for (int j = 0; j < 2; ++j) {
      union { f32x8 v; float f[8]; } c; c.v = acc[i][j];
      const int n = col0 + j * 16 + nl;
      for (int v = 0; v < 8; ++v) {
        const int m = row0 + i * 16 + g * 8 + v;
        const float gn = c.f[v] + GIn_t[(size_t)m * ldGI + n];
        const float nv = tanhf(gn);
        const float z  = z_buf[m * H + n];
        const float hp = h_f[m * H + n];
        const float hn = (1.0f - z) * nv + z * hp;
        h_f[m * H + n]  = hn;
        h_bf[m * H + n] = (__bf16)hn;
        if (y_bf) y_bf[((size_t)m * T + t) * H + n] = (__bf16)hn;
        out[(size_t)m * outW + out_off + n] += mask[(size_t)m * T + t] * hn;
      }
    }
}

// ---------------------------------------------------------------------------
extern "C" void kernel_launch(void* const* d_in, const int* in_sizes, int n_in,
                              void* d_out, int out_size, void* d_ws, size_t ws_size,
                              hipStream_t stream) {
  (void)in_sizes; (void)n_in;
  constexpr int B = 128, T = 128, I = 512;
  constexpr int H[3] = {512, 1024, 2048};
  constexpr int OUTW = 512 + 1024 + 2048;

  const float* x    = (const float*)d_in[0];
  const float* mask = (const float*)d_in[1];
  const float* Wih[3]  = {(const float*)d_in[2], (const float*)d_in[5], (const float*)d_in[8]};
  const float* Whh[3]  = {(const float*)d_in[3], (const float*)d_in[6], (const float*)d_in[9]};
  const float* bias[3] = {(const float*)d_in[4], (const float*)d_in[7], (const float*)d_in[10]};
  float* out = (float*)d_out;

  // ---- workspace bump allocator -------------------------------------------
  char* p = (char*)d_ws;
  size_t used = 0;
  auto alloc = [&](size_t bytes) -> void* {
    bytes = (bytes + 255) & ~(size_t)255;
    void* r = p + used;
    used += bytes;
    return r;
  };
  __bf16* x_bf   = (__bf16*)alloc((size_t)B * T * I * 2);
  __bf16* y_bf0  = (__bf16*)alloc((size_t)B * T * H[0] * 2);
  __bf16* y_bf1  = (__bf16*)alloc((size_t)B * T * H[1] * 2);
  __bf16* wih_bf[3], * whh_bf[3];
  const int Kin[3] = {I, H[0], H[1]};
  for (int l = 0; l < 3; ++l) {
    wih_bf[l] = (__bf16*)alloc((size_t)3 * H[l] * Kin[l] * 2);
    whh_bf[l] = (__bf16*)alloc((size_t)3 * H[l] * H[l] * 2);
  }
  float*  GI    = (float*) alloc((size_t)B * T * 3 * H[2] * 4);  // reused per layer
  float*  h_f   = (float*) alloc((size_t)B * H[2] * 4);
  __bf16* h_bf  = (__bf16*)alloc((size_t)B * H[2] * 2);
  float*  z_buf = (float*) alloc((size_t)B * H[2] * 4);
  __bf16* rh_bf = (__bf16*)alloc((size_t)B * H[2] * 2);

  hipMemsetAsync(d_out, 0, sizeof(float) * (size_t)out_size, stream);
  if (used > ws_size) return;  // out stays zero; deterministic

  // ---- one-time (per launch) bf16 casts -----------------------------------
  auto cast = [&](const float* s, __bf16* d, size_t n) {
    int n4 = (int)(n / 4);
    cast_f32_to_bf16_x4<<<(n4 + 255) / 256, 256, 0, stream>>>(s, d, n4);
  };
  cast(x, x_bf, (size_t)B * T * I);
  for (int l = 0; l < 3; ++l) {
    cast(Wih[l], wih_bf[l], (size_t)3 * H[l] * Kin[l]);
    cast(Whh[l], whh_bf[l], (size_t)3 * H[l] * H[l]);
  }

  // ---- layers --------------------------------------------------------------
  const __bf16* Xl[3] = {x_bf, y_bf0, y_bf1};
  __bf16* Yl[3] = {y_bf0, y_bf1, nullptr};
  int out_off = 0;
  for (int l = 0; l < 3; ++l) {
    const int Hh = H[l], K = Kin[l], N3 = 3 * Hh;
    hipMemsetAsync(h_f,  0, (size_t)B * Hh * 4, stream);
    hipMemsetAsync(h_bf, 0, (size_t)B * Hh * 2, stream);

    // Parallel input projection over all timesteps: GI = X @ Wih^T + b
    dim3 ggrid(N3 / 128, (B * T) / 128);
    gi_gemm_kernel<<<ggrid, 256, 0, stream>>>(Xl[l], wih_bf[l], bias[l], GI, N3, K);

    // Sequential recurrence
    const int ldGI = T * N3;
    for (int t = 0; t < T; ++t) {
      const float* GI_t  = GI + (size_t)t * N3;
      const float* GIn_t = GI_t + 2 * Hh;
      zr_step_kernel<<<dim3((2 * Hh) / 64), 256, 0, stream>>>(
          h_bf, h_f, whh_bf[l], GI_t, ldGI, z_buf, rh_bf, Hh);
      n_step_kernel<<<dim3(Hh / 64), 256, 0, stream>>>(
          rh_bf, whh_bf[l] + (size_t)2 * Hh * Hh, GIn_t, ldGI, z_buf,
          h_f, h_bf, Yl[l], out, mask, t, T, Hh, out_off, OUTW);
    }
    out_off += Hh;
  }
}
